// NT_Xent_26293789786309
// MI455X (gfx1250) — matrix operand order repaired and checked by reading, not compiled, not run on hardware
//
#include <hip/hip_runtime.h>
#include <hip/hip_bf16.h>
#include <math.h>

typedef _Float16 half_t;
typedef __attribute__((ext_vector_type(8)))  _Float16 v8h;
typedef __attribute__((ext_vector_type(16))) _Float16 v16h;
typedef __attribute__((ext_vector_type(8)))  float    v8f;
typedef __attribute__((ext_vector_type(4)))  int      v4i;

#define N2      8192   // 2*BATCH
#define DIMC    128
#define CCHUNK  8      // column chunks (grid.y)
#define PANEL   64     // columns per LDS panel (double-buffered)
#define NPANELS ((N2 / CCHUNK) / PANEL)   // 16
#define LSTRIDE 136    // padded LDS row stride in halves (conflict-free b128)

// exp(sim-2) with sim = 2*dot  ->  exp2(dot*K1 - K1),  K1 = 2*log2(e)
#define K1F  2.8853900817779268f
#define LN2F 0.6931471805599453f

#if defined(__HIP_DEVICE_COMPILE__) && __has_builtin(__builtin_amdgcn_exp2f)
#define EXP2(x) __builtin_amdgcn_exp2f(x)
#else
#define EXP2(x) exp2f(x)
#endif
#if defined(__HIP_DEVICE_COMPILE__) && __has_builtin(__builtin_amdgcn_logf)
#define LOG2(x) __builtin_amdgcn_logf(x)
#else
#define LOG2(x) log2f(x)
#endif

// gfx1250 async global->LDS staging (ASYNCcnt) when the toolchain exposes it
#if defined(__HIP_DEVICE_COMPILE__) && \
    __has_builtin(__builtin_amdgcn_global_load_async_to_lds_b128) && \
    __has_builtin(__builtin_amdgcn_s_wait_asynccnt)
#define HAVE_ASYNC_LDS 1
#define GLD_ASYNC_B128(g, l)                                              \
    __builtin_amdgcn_global_load_async_to_lds_b128(                       \
        (__attribute__((address_space(1))) v4i*)(g),                      \
        (__attribute__((address_space(3))) v4i*)(l), 0, 0)
#define WAIT_ASYNC0() __builtin_amdgcn_s_wait_asynccnt(0)
#else
#define HAVE_ASYNC_LDS 0
#endif

__device__ __forceinline__ v16h ldv16(const half_t* p) {
    v8h lo = *(const v8h*)(p);
    v8h hi = *(const v8h*)(p + 8);
    return __builtin_shufflevector(lo, hi, 0, 1, 2, 3, 4, 5, 6, 7,
                                   8, 9, 10, 11, 12, 13, 14, 15);
}

// ---------------------------------------------------------------------------
// Kernel 1: row-normalize z = [z_i; z_j] -> f16 zn in workspace
// ---------------------------------------------------------------------------
__global__ __launch_bounds__(128) void normalize_k(const float* __restrict__ zi,
                                                   const float* __restrict__ zj,
                                                   half_t* __restrict__ zn) {
    const int row = blockIdx.x;
    const int t   = threadIdx.x;
    const float* src = (row < N2 / 2) ? (zi + (size_t)row * DIMC)
                                      : (zj + (size_t)(row - N2 / 2) * DIMC);
    float v  = src[t];
    float ss = v * v;
#pragma unroll
    for (int m = 1; m < 32; m <<= 1) ss += __shfl_xor(ss, m, 32);
    __shared__ float s[4];
    if ((t & 31) == 0) s[t >> 5] = ss;
    __syncthreads();
    float tot = s[0] + s[1] + s[2] + s[3];
    float inv = 1.0f / fmaxf(sqrtf(tot), 1e-8f);
    zn[(size_t)row * DIMC + t] = (half_t)(v * inv);
}

// ---------------------------------------------------------------------------
// Kernel 2: fused zn @ zn^T with Sum exp(sim-2) per (row, column-chunk).
// grid (64, 8): blockIdx.x = 128-row M-stripe, blockIdx.y = 1024-col chunk.
// 8 waves/block, 1 wave = 16 rows. Double-buffered 64-col LDS panels, two
// barriers per panel; async global->LDS when available. Branch-free hot loop:
// per tile-pair 16 ds_load_b128 -> 8 WMMA (2 independent acc chains) -> exp.
// ---------------------------------------------------------------------------
__global__ __launch_bounds__(256) void ntxent_main(const half_t* __restrict__ zn,
                                                   float* __restrict__ partial) {
    __shared__ __align__(16) half_t tile[2][PANEL * LSTRIDE];   // 34816 B

    const int tid   = threadIdx.x;
    const int wave  = tid >> 5;
    const int lane  = tid & 31;
    const int g     = lane >> 4;    // half-wave group
    const int cr    = lane & 15;    // row-in-tile (A) / col-in-tile (B/C)
    const int mbase = blockIdx.x * 128 + wave * 16;
    const int cbase = blockIdx.y * (N2 / CCHUNK);

    // ---- A tile (16 rows x 128 K) in WMMA A-layout, register-resident ----
    // lane L holds row (L&15); per K32 chunk, halves K = {16g..16g+15}
    v16h a[4];
    {
        const half_t* rp = zn + (size_t)(mbase + cr) * DIMC;
#pragma unroll
        for (int kc = 0; kc < 4; ++kc) {
            v8h lo = *(const v8h*)(rp + kc * 32 + g * 16);
            v8h hi = *(const v8h*)(rp + kc * 32 + g * 16 + 8);
            a[kc] = __builtin_shufflevector(lo, hi, 0, 1, 2, 3, 4, 5, 6, 7,
                                            8, 9, 10, 11, 12, 13, 14, 15);
        }
    }

    float acc[8];
#pragma unroll
    for (int v = 0; v < 8; ++v) acc[v] = 0.f;

    // Staging map: thread t moves 64 B of panel row (t>>2), 4 x b128
    const int srow = tid >> 2;
    const int soff = (tid & 3) * 32;                       // halves
    const half_t* sptr = zn + ((size_t)cbase + srow) * DIMC + soff;

#if HAVE_ASYNC_LDS
    // ---- async global->LDS staging, ASYNCcnt-tracked ----
    {
        half_t* l0 = &tile[0][srow * LSTRIDE + soff];
#pragma unroll
        for (int j = 0; j < 4; ++j)
            GLD_ASYNC_B128(sptr + j * 8, l0 + j * 8);
    }
#else
    v8h pre[4];
#pragma unroll
    for (int j = 0; j < 4; ++j) pre[j] = *(const v8h*)(sptr + j * 8);
#endif

#pragma unroll 1
    for (int p = 0; p < NPANELS; ++p) {
        const half_t* cur = &tile[p & 1][0];

#if HAVE_ASYNC_LDS
        WAIT_ASYNC0();                 // panel p landed in LDS (this wave's part)
        __syncthreads();               // all waves' parts in; prev reads done
        if (p < NPANELS - 1) {         // overlap copy of p+1 with compute of p
            const half_t* np = sptr + (size_t)(p + 1) * PANEL * DIMC;
            half_t* nl = &tile[(p + 1) & 1][srow * LSTRIDE + soff];
#pragma unroll
            for (int j = 0; j < 4; ++j)
                GLD_ASYNC_B128(np + j * 8, nl + j * 8);
        }
#else
        {
            half_t* lp = &tile[p & 1][srow * LSTRIDE + soff];
#pragma unroll
            for (int j = 0; j < 4; ++j) *(v8h*)(lp + j * 8) = pre[j];
        }
        if (p < NPANELS - 1) {
            const half_t* np = sptr + (size_t)(p + 1) * PANEL * DIMC;
#pragma unroll
            for (int j = 0; j < 4; ++j) pre[j] = *(const v8h*)(np + j * 8);
        }
        __syncthreads();               // panel p staged; prev reads done
#endif

        // ---- 4 column tiles as 2 pairs: preload B regs, dual WMMA chains ----
#pragma unroll
        for (int pr2 = 0; pr2 < 2; ++pr2) {
            const half_t* b0p = cur + (pr2 * 32 + cr) * LSTRIDE + g * 16;
            const half_t* b1p = b0p + 16 * LSTRIDE;
            v16h b0[4], b1[4];
#pragma unroll
            for (int kc = 0; kc < 4; ++kc) {
                b0[kc] = ldv16(b0p + kc * 32);
                b1[kc] = ldv16(b1p + kc * 32);
            }
            v8f c0, c1;
#pragma unroll
            for (int i = 0; i < 8; ++i) { c0[i] = 0.f; c1[i] = 0.f; }
#pragma unroll
            for (int kc = 0; kc < 4; ++kc) {
                c0 = __builtin_amdgcn_wmma_f32_16x16x32_f16(
                         false, a[kc], false, b0[kc], (short)0, c0, false, false);
                c1 = __builtin_amdgcn_wmma_f32_16x16x32_f16(
                         false, a[kc], false, b1[kc], (short)0, c1, false, false);
            }
            // C layout: VGPR v, lane L -> row (v + 8g), col (L&15)
#pragma unroll
            for (int v = 0; v < 8; ++v) {
                acc[v] += EXP2(__builtin_fmaf(c0[v], K1F, -K1F));
                acc[v] += EXP2(__builtin_fmaf(c1[v], K1F, -K1F));
            }
        }
        __syncthreads();               // compute done before next stage reuses buf
    }

    // reduce across the 16 lanes of each half-wave (same rows per group)
#pragma unroll
    for (int v = 0; v < 8; ++v) {
#pragma unroll
        for (int m = 1; m < 16; m <<= 1)
            acc[v] += __shfl_xor(acc[v], m, 32);
    }
    if (cr == 0) {
        float* pp = partial + (size_t)blockIdx.y * N2 + mbase + 8 * g;
#pragma unroll
        for (int v = 0; v < 8; ++v) pp[v] = acc[v];
    }
}

// ---------------------------------------------------------------------------
// Kernel 3: per-row finalize. self/pos dots from zn, combine chunk partials,
// contrib[r] = 2 + ln(Sum_{c!=r} exp(sim-2)) - pos.
// ---------------------------------------------------------------------------
__global__ __launch_bounds__(128) void finalize_rows(const half_t* __restrict__ zn,
                                                     const float* __restrict__ partial,
                                                     float* __restrict__ contrib) {
    const int r  = blockIdx.x;
    const int t  = threadIdx.x;
    const int pr = (r + N2 / 2) & (N2 - 1);                // positive pair row
    float av = (float)zn[(size_t)r  * DIMC + t];
    float bv = (float)zn[(size_t)pr * DIMC + t];
    float dp = av * bv;    // pos dot
    float dq = av * av;    // self dot
#pragma unroll
    for (int m = 1; m < 32; m <<= 1) {
        dp += __shfl_xor(dp, m, 32);
        dq += __shfl_xor(dq, m, 32);
    }
    __shared__ float sp[4], sq[4];
    if ((t & 31) == 0) { sp[t >> 5] = dp; sq[t >> 5] = dq; }
    __syncthreads();
    if (t == 0) {
        float dpt = sp[0] + sp[1] + sp[2] + sp[3];
        float dqt = sq[0] + sq[1] + sq[2] + sq[3];
        float tot = 0.f;
#pragma unroll
        for (int cc = 0; cc < CCHUNK; ++cc) tot += partial[(size_t)cc * N2 + r];
        tot -= EXP2(__builtin_fmaf(dqt, K1F, -K1F));       // drop c == r
        contrib[r] = 2.0f + LN2F * LOG2(tot) - 2.0f * dpt; // lse - pos
    }
}

// ---------------------------------------------------------------------------
// Kernel 4: deterministic fixed-order reduction of 8192 row contributions
// ---------------------------------------------------------------------------
__global__ __launch_bounds__(1024) void reduce_all(const float* __restrict__ contrib,
                                                   float* __restrict__ out) {
    __shared__ float s[1024];
    const int t = threadIdx.x;
    float v = 0.f;
#pragma unroll
    for (int i = 0; i < N2 / 1024; ++i) v += contrib[t + i * 1024];
    s[t] = v;
    __syncthreads();
    for (int off = 512; off > 0; off >>= 1) {
        if (t < off) s[t] += s[t + off];
        __syncthreads();
    }
    if (t == 0) out[0] = s[0] * (1.0f / (float)N2);
}

// ---------------------------------------------------------------------------
extern "C" void kernel_launch(void* const* d_in, const int* in_sizes, int n_in,
                              void* d_out, int out_size, void* d_ws, size_t ws_size,
                              hipStream_t stream) {
    const float* zi = (const float*)d_in[0];
    const float* zj = (const float*)d_in[1];
    // d_in[2] (mask) unused: mask structure derived analytically.

    half_t* zn      = (half_t*)d_ws;                                   // 2 MB
    float*  partial = (float*)((char*)d_ws + (size_t)N2 * DIMC * 2);   // 256 KB
    float*  contrib = partial + (size_t)CCHUNK * N2;                   // 32 KB
    float*  out     = (float*)d_out;

    normalize_k<<<N2, DIMC, 0, stream>>>(zi, zj, zn);
    ntxent_main<<<dim3(N2 / 128, CCHUNK), 256, 0, stream>>>(zn, partial);
    finalize_rows<<<N2, DIMC, 0, stream>>>(zn, partial, contrib);
    reduce_all<<<1, 1024, 0, stream>>>(contrib, out);
}
